// Graphpool_81827716924049
// MI455X (gfx1250) — matrix operand order repaired and checked by reading, not compiled, not run on hardware
//
#include <hip/hip_runtime.h>

typedef __attribute__((ext_vector_type(16))) __bf16 v16bf;
typedef __attribute__((ext_vector_type(8)))  float  v8f;

#define BB 32
#define NN 1000
#define NP 1024
#define DD 64
#define NT 63          // 16-row dst tiles covering 1008 >= 1000
#define NEGV (-1e8f)

__device__ __forceinline__ unsigned short f2bf(float f) {
    union { float f; unsigned u; } c; c.f = f;
    unsigned u = c.u;
    unsigned r = (u + 0x7FFFu + ((u >> 16) & 1u)) >> 16;   // RNE
    return (unsigned short)r;
}

union Frag { v16bf v; uint4 q[2]; };
union U4   { uint4 q; unsigned short u[8]; };

// ---------------------------------------------------------------- features
__global__ void k_build_x(const int* __restrict__ op_idx, const float* __restrict__ vf,
                          const float* __restrict__ tp_tab, const float* __restrict__ cs_tab,
                          float* __restrict__ x) {
    int idx = blockIdx.x * 256 + threadIdx.x;       // b*1024+n
    if (idx >= BB * NP) return;
    int n = idx & (NP - 1), b = idx >> 10;
    float row[16];
#pragma unroll
    for (int f = 0; f < 16; ++f) row[f] = 0.f;
    if (n < NN) {
        int g = b * NN + n;
        row[0] = cs_tab[n * 4 + 0]; row[1] = cs_tab[n * 4 + 1];
        row[2] = cs_tab[n * 4 + 2]; row[3] = cs_tab[n * 4 + 3];
        int op = op_idx[g];
        row[4] = tp_tab[op * 2 + 0]; row[5] = tp_tab[op * 2 + 1];
#pragma unroll
        for (int k = 0; k < 9; ++k) row[6 + k] = vf[(size_t)g * 9 + k];
    }
#pragma unroll
    for (int f = 0; f < 16; ++f) x[(size_t)idx * 16 + f] = row[f];
}

// ------------------------------------------------- h = [x|rep] @ W, + hd/hs
__global__ void k_gemm_h(const float* __restrict__ x, int f1,
                         const float* __restrict__ rep, int f2,
                         const float* __restrict__ W,
                         const float* __restrict__ asrc, const float* __restrict__ adst,
                         float* __restrict__ h, unsigned short* __restrict__ hb,
                         float* __restrict__ hs, float* __restrict__ hd) {
    int row = blockIdx.x;           // b*1024+n
    int d = threadIdx.x;            // 0..63
    __shared__ float sx[80];
    __shared__ float red[64];
    if (d < f1) sx[d] = x[(size_t)row * 16 + d];
    if (f2 && d < f2) sx[f1 + d] = rep[(size_t)row * 64 + d];
    __syncthreads();
    int F = f1 + f2;
    float acc = 0.f;
    for (int f = 0; f < F; ++f) acc += sx[f] * W[f * 64 + d];
    h[(size_t)row * 64 + d] = acc;
    hb[(size_t)row * 64 + d] = f2bf(acc);
    red[d] = acc * adst[d];
    __syncthreads();
    for (int s = 32; s > 0; s >>= 1) { if (d < s) red[d] += red[d + s]; __syncthreads(); }
    if (d == 0) hd[row] = red[0];
    __syncthreads();
    red[d] = acc * asrc[d];
    __syncthreads();
    for (int s = 32; s > 0; s >>= 1) { if (d < s) red[d] += red[d + s]; __syncthreads(); }
    if (d == 0) hs[row] = red[0];
}

// ------------------- fused GAT: masked logits + online softmax + WMMA agg
__global__ void k_gat(const unsigned short* __restrict__ hb,
                      const float* __restrict__ hd, const float* __restrict__ hs,
                      const unsigned char* __restrict__ adj,
                      const float* __restrict__ dist,
                      const float* __restrict__ wd_p,
                      float* __restrict__ out) {
    const float wd = wd_p[0];
    int b  = blockIdx.x / NT;
    int i0 = (blockIdx.x % NT) * 16;
    int t = threadIdx.x;
    int lane = t & 31, wave = t >> 5;
    int hlf = lane >> 4, col = lane & 15;

    __shared__ float s_hd[16], s_hs[32];
    __shared__ float s_e[16][33];
    __shared__ float s_m[16], s_scale[16], s_l[16], s_lrow[16];
    __shared__ __align__(16) unsigned short s_p[16 * 40];    // row stride 40 (80B)
    __shared__ __align__(16) unsigned short s_hbT[64 * 40];  // transposed [d][j]

    if (t < 16) {
        int i = i0 + t;
        s_hd[t] = (i < NN) ? hd[b * NP + i] : 0.f;
        s_m[t] = -1e30f; s_l[t] = 0.f;
    }

    v8f acc = {};
    int ii = t & 15;
    int jb = (t >> 4) & 7;
    int dblk = t & 7, jp = t >> 3;     // staging transpose coords

    for (int jt = 0; jt < 32; ++jt) {
        int j0 = jt * 32;
        __syncthreads();                         // protect LDS from prev iter
        if (t < 32) { int j = j0 + t; s_hs[t] = (j < NN) ? hs[b * NP + j] : 0.f; }
        {   // stage 32x64 bf16 h-tile, transposed to s_hbT[d][j]
            U4 r0, r1;
            r0.q = *(const uint4*)(hb + (size_t)(b * NP + j0 + 2 * jp) * 64 + dblk * 8);
            r1.q = *(const uint4*)(hb + (size_t)(b * NP + j0 + 2 * jp + 1) * 64 + dblk * 8);
#pragma unroll
            for (int u = 0; u < 8; ++u) {
                int d = dblk * 8 + u;
                unsigned v32 = (unsigned)r0.u[u] | ((unsigned)r1.u[u] << 16);
                *(unsigned*)&s_hbT[d * 40 + 2 * jp] = v32;
            }
        }
        if (t < 16) s_lrow[t] = 0.f;
        __syncthreads();
        // logits e[ii][jj] = leaky_relu(hd_i + hs_j + wd*dist[j,i]) masked
#pragma unroll
        for (int q = 0; q < 4; ++q) {
            int jj = jb + 8 * q;
            int i = i0 + ii, j = j0 + jj;
            float e = -1e30f;
            if (i < NN && j < NN &&
                adj[(size_t)b * NN * NN + (size_t)j * NN + i]) {
                float v = s_hd[ii] + s_hs[jj] + wd * dist[(size_t)j * NN + i];
                e = v > 0.f ? v : 0.2f * v;
            }
            s_e[ii][jj] = e;
        }
        __syncthreads();
        if (t < 16) {                            // online-softmax row state
            float m = s_m[t], mt = -1e30f;
            for (int jj = 0; jj < 32; ++jj) mt = fmaxf(mt, s_e[t][jj]);
            float mn = fmaxf(m, mt);
            s_scale[t] = __expf(m - mn);
            s_m[t] = mn;
        }
        __syncthreads();
        {
            float psum = 0.f;
#pragma unroll
            for (int q = 0; q < 4; ++q) {
                int jj = jb + 8 * q;
                float e = s_e[ii][jj];
                float p = (e > -1e29f) ? __expf(e - s_m[ii]) : 0.f;
                s_p[ii * 40 + jj] = f2bf(p);
                psum += p;
            }
            if (psum != 0.f) atomicAdd(&s_lrow[ii], psum);
        }
        __syncthreads();
        if (t < 16) s_l[t] = s_l[t] * s_scale[t] + s_lrow[t];
        // rescale accumulator (C fragment rows r, r+8 per lane half)
#pragma unroll
        for (int r = 0; r < 8; ++r) acc[r] *= s_scale[r + hlf * 8];
        // fragments: two b128 LDS loads each (contiguous k = 16*half + s, same order for A and B)
        Frag af, bf2;
        int arow = lane & 15;
        af.q[0] = *(const uint4*)&s_p[arow * 40 + hlf * 16];
        af.q[1] = *(const uint4*)&s_p[arow * 40 + hlf * 16 + 8];
        int nc = wave * 16 + col;
        bf2.q[0] = *(const uint4*)&s_hbT[nc * 40 + hlf * 16];
        bf2.q[1] = *(const uint4*)&s_hbT[nc * 40 + hlf * 16 + 8];
        acc = __builtin_amdgcn_wmma_f32_16x16x32_bf16(
                  false, af.v, false, bf2.v, (short)0, acc, false, false);
    }
    __syncthreads();
#pragma unroll
    for (int r = 0; r < 8; ++r) {
        int row = r + hlf * 8;
        int i = i0 + row;
        if (i < NN) {
            float l = s_l[row];
            float v = (l > 0.f) ? acc[r] / l : 0.f;     // isolated node -> 0
            v = v > 0.f ? v : (__expf(v) - 1.f);        // elu
            out[((size_t)b * NP + i) * 64 + wave * 16 + col] = v;
        }
    }
}

// ---------------------------------------------------------------- sag (bf16)
__global__ void k_sag(const int* __restrict__ time_idx, const int* __restrict__ op_idx,
                      const float* __restrict__ vf,
                      const float* __restrict__ time_tab, const float* __restrict__ tp_tab,
                      const float* __restrict__ cs_tab,
                      const float* __restrict__ comp, const float* __restrict__ coop,
                      unsigned short* __restrict__ sag) {
    size_t idx = (size_t)blockIdx.x * 256 + threadIdx.x;      // B*1024*160
    if (idx >= (size_t)BB * NP * 160) return;
    int f = (int)(idx % 160);
    size_t rn = idx / 160;
    int n = (int)(rn & (NP - 1)), b = (int)(rn >> 10);
    float v = 0.f;
    if (n < NN && f < 147) {
        int g = b * NN + n;
        if (f < 4)       v = time_tab[time_idx[g] * 4 + f];
        else if (f < 8)  v = cs_tab[n * 4 + (f - 4)];
        else if (f < 10) v = tp_tab[op_idx[g] * 2 + (f - 8)];
        else if (f < 19) v = vf[(size_t)g * 9 + (f - 10)];
        else if (f < 83) v = comp[rn * 64 + (f - 19)];
        else             v = coop[rn * 64 + (f - 83)];
    }
    sag[idx] = f2bf(v);
}

__global__ void k_wpconv(const float* __restrict__ Wp, unsigned short* __restrict__ Wb) {
    int idx = blockIdx.x * 256 + threadIdx.x;    // 160*256
    if (idx >= 160 * 256) return;
    int r = idx >> 8, c = idx & 255;
    Wb[idx] = (r < 147) ? f2bf(Wp[r * 256 + c]) : (unsigned short)0;
}

// ------------------------------------------- x_p = sag @ Wp + bp (WMMA GEMM)
__global__ void k_proj(const unsigned short* __restrict__ A,    // (32768,160)
                       const unsigned short* __restrict__ Bm,   // (160,256)
                       const float* __restrict__ bp,
                       float* __restrict__ xp) {
    int row0 = blockIdx.x * 16;
    int col0 = blockIdx.y * 64;
    int t = threadIdx.x, lane = t & 31, wave = t >> 5;
    int hlf = lane >> 4, col = lane & 15;
    __shared__ __align__(16) unsigned short sA[16 * 40];
    __shared__ __align__(16) unsigned short sBT[64 * 40];   // transposed [col][k]
    v8f acc = {};
    int cb = t & 7, kp = t >> 3;       // staging transpose coords
#pragma unroll
    for (int kt = 0; kt < 5; ++kt) {
        __syncthreads();
        if (t < 64) {          // A tile: 16 rows x 32 k, direct uint4 copy
            int r = t >> 2, c4 = t & 3;
            uint4 a = *(const uint4*)(A + (size_t)(row0 + r) * 160 + kt * 32 + c4 * 8);
            *(uint4*)&sA[r * 40 + c4 * 8] = a;
        }
        {   // B tile: 32 k x 64 cols, transposed to sBT[col][k]
            U4 r0, r1;
            r0.q = *(const uint4*)(Bm + (size_t)(kt * 32 + 2 * kp) * 256 + col0 + cb * 8);
            r1.q = *(const uint4*)(Bm + (size_t)(kt * 32 + 2 * kp + 1) * 256 + col0 + cb * 8);
#pragma unroll
            for (int u = 0; u < 8; ++u) {
                int c = cb * 8 + u;
                unsigned v32 = (unsigned)r0.u[u] | ((unsigned)r1.u[u] << 16);
                *(unsigned*)&sBT[c * 40 + 2 * kp] = v32;
            }
        }
        __syncthreads();
        Frag af, bf2;
        int arow = lane & 15;
        af.q[0] = *(const uint4*)&sA[arow * 40 + hlf * 16];
        af.q[1] = *(const uint4*)&sA[arow * 40 + hlf * 16 + 8];
        int nc = wave * 16 + col;
        bf2.q[0] = *(const uint4*)&sBT[nc * 40 + hlf * 16];
        bf2.q[1] = *(const uint4*)&sBT[nc * 40 + hlf * 16 + 8];
        acc = __builtin_amdgcn_wmma_f32_16x16x32_bf16(
                  false, af.v, false, bf2.v, (short)0, acc, false, false);
    }
#pragma unroll
    for (int r = 0; r < 8; ++r) {
        int row = row0 + r + hlf * 8;
        int c = col0 + wave * 16 + col;
        xp[(size_t)row * 256 + c] = acc[r] + bp[c];
    }
}

// ---------------------------------------------------------------- scores
__global__ void k_scores(const float* __restrict__ xp, const float* __restrict__ vpd,
                         const float* __restrict__ vpf,
                         float* __restrict__ sd, float* __restrict__ sf) {
    int row = blockIdx.x; int h = threadIdx.x;
    float v = xp[(size_t)row * 256 + h];
    __shared__ float r1[256], r2[256];
    r1[h] = v * vpd[h]; r2[h] = v * vpf[h];
    __syncthreads();
    for (int s = 128; s > 0; s >>= 1) {
        if (h < s) { r1[h] += r1[h + s]; r2[h] += r2[h + s]; }
        __syncthreads();
    }
    if (h == 0) { sd[row] = r1[0]; sf[row] = r2[0]; }
}

// --------------------------- per (b,pool): smax + 500th-largest via bitonic
__global__ void k_topk(const float* __restrict__ sd, const float* __restrict__ sf,
                       const unsigned char* __restrict__ mark,
                       float* __restrict__ smax_out, float* __restrict__ thr_out) {
    int b = blockIdx.x >> 1; int type = blockIdx.x & 1;   // 0=f(~mark) 1=d(mark)
    const float* s = type ? sd : sf;
    int t = threadIdx.x;
    __shared__ float d[1024];
    __shared__ float red[1024];
    float v = (t < NN) ? s[b * NP + t] : -3.0e38f;
    red[t] = v;
    __syncthreads();
    for (int st = 512; st > 0; st >>= 1) {
        if (t < st) red[t] = fmaxf(red[t], red[t + st]);
        __syncthreads();
    }
    float smax = red[0];
    bool valid = false;
    if (t < NN) { unsigned char m = mark[b * NN + t]; valid = type ? (m != 0) : (m == 0); }
    float sc = (t < NN) ? (valid ? v - smax : NEGV) : -3.0e38f;
    d[t] = sc;
    __syncthreads();
    for (int k = 2; k <= 1024; k <<= 1)
        for (int j = k >> 1; j > 0; j >>= 1) {
            int ixj = t ^ j;
            if (ixj > t) {
                bool dir = ((t & k) == 0);               // descending overall
                float a = d[t], c = d[ixj];
                if ((dir && a < c) || (!dir && a > c)) { d[t] = c; d[ixj] = a; }
            }
            __syncthreads();
        }
    if (t == 0) { smax_out[blockIdx.x] = smax; thr_out[blockIdx.x] = d[499]; }
}

// ------------------------------------------------ gated pool: softmax+sum/max
__global__ void k_pool(const float* __restrict__ xp, const float* __restrict__ sd,
                       const float* __restrict__ sf, const unsigned char* __restrict__ mark,
                       const float* __restrict__ smax_in, const float* __restrict__ thr_in,
                       float* __restrict__ out) {
    int b = blockIdx.x >> 1; int type = blockIdx.x & 1;
    const float* s = type ? sd : sf;
    float smax = smax_in[blockIdx.x], thr = thr_in[blockIdx.x];
    int t = threadIdx.x;
    __shared__ float w[NN];
    __shared__ float red[256];
    float lmx = -3e38f;
    for (int n = t; n < NN; n += 256) {
        unsigned char m = mark[b * NN + n];
        bool valid = type ? (m != 0) : (m == 0);
        float sc = valid ? s[b * NP + n] - smax : NEGV;
        if (sc < thr) sc = NEGV;
        w[n] = sc;
        lmx = fmaxf(lmx, sc);
    }
    red[t] = lmx; __syncthreads();
    for (int st = 128; st > 0; st >>= 1) {
        if (t < st) red[t] = fmaxf(red[t], red[t + st]);
        __syncthreads();
    }
    float mx = red[0];
    __syncthreads();
    float lsum = 0.f;
    for (int n = t; n < NN; n += 256) { float e = __expf(w[n] - mx); w[n] = e; lsum += e; }
    red[t] = lsum; __syncthreads();
    for (int st = 128; st > 0; st >>= 1) {
        if (t < st) red[t] += red[t + st];
        __syncthreads();
    }
    float inv = 1.f / red[0];
    __syncthreads();
    for (int n = t; n < NN; n += 256) w[n] *= inv;
    __syncthreads();
    float sum = 0.f, mxv = -3e38f;
    for (int n = 0; n < NN; ++n) {
        float v = xp[((size_t)b * NP + n) * 256 + t] * w[n];
        sum += v; mxv = fmaxf(mxv, v);
    }
    int base = type ? 512 : 0;        // out row: [sum_f, max_f, sum_d, max_d]
    out[(size_t)b * 1024 + base + t] = sum;
    out[(size_t)b * 1024 + base + 256 + t] = mxv;
}

extern "C" void kernel_launch(void* const* d_in, const int* in_sizes, int n_in,
                              void* d_out, int out_size, void* d_ws, size_t ws_size,
                              hipStream_t stream) {
    (void)in_sizes; (void)n_in; (void)out_size; (void)ws_size;
    const int*   time_idx = (const int*)d_in[0];
    const int*   op_idx   = (const int*)d_in[1];
    const float* vf       = (const float*)d_in[2];
    const unsigned char* mark = (const unsigned char*)d_in[3];
    const unsigned char* adjc = (const unsigned char*)d_in[4];
    const unsigned char* adjo = (const unsigned char*)d_in[5];
    const float* dist     = (const float*)d_in[6];
    const float* time_tab = (const float*)d_in[7];
    const float* tp_tab   = (const float*)d_in[8];
    const float* cs_tab   = (const float*)d_in[9];
    const float* Wc      = (const float*)d_in[10];
    const float* asrc_c  = (const float*)d_in[11];
    const float* adst_c  = (const float*)d_in[12];
    const float* wd_c    = (const float*)d_in[13];
    const float* Wco     = (const float*)d_in[14];
    const float* asrc_co = (const float*)d_in[15];
    const float* adst_co = (const float*)d_in[16];
    const float* wd_co   = (const float*)d_in[17];
    const float* Wp  = (const float*)d_in[18];
    const float* bp  = (const float*)d_in[19];
    const float* vpd = (const float*)d_in[20];
    const float* vpf = (const float*)d_in[21];
    float* out = (float*)d_out;

    char* ws = (char*)d_ws;
    size_t off = 0;
    auto alloc = [&](size_t bytes) -> char* {
        char* p = ws + off;
        off = (off + bytes + 255) & ~(size_t)255;
        return p;
    };
    float* x    = (float*)alloc((size_t)BB * NP * 16 * 4);
    float* h1   = (float*)alloc((size_t)BB * NP * 64 * 4);
    unsigned short* hb1 = (unsigned short*)alloc((size_t)BB * NP * 64 * 2);
    float* hd1  = (float*)alloc((size_t)BB * NP * 4);
    float* hs1  = (float*)alloc((size_t)BB * NP * 4);
    float* comp = (float*)alloc((size_t)BB * NP * 64 * 4);
    float* h2   = (float*)alloc((size_t)BB * NP * 64 * 4);
    unsigned short* hb2 = (unsigned short*)alloc((size_t)BB * NP * 64 * 2);
    float* hd2  = (float*)alloc((size_t)BB * NP * 4);
    float* hs2  = (float*)alloc((size_t)BB * NP * 4);
    float* coop = (float*)alloc((size_t)BB * NP * 64 * 4);
    unsigned short* sag = (unsigned short*)alloc((size_t)BB * NP * 160 * 2);
    unsigned short* Wb  = (unsigned short*)alloc((size_t)160 * 256 * 2);
    float* xp   = (float*)alloc((size_t)BB * NP * 256 * 4);
    float* sd   = (float*)alloc((size_t)BB * NP * 4);
    float* sf   = (float*)alloc((size_t)BB * NP * 4);
    float* smax = (float*)alloc(64 * 4);
    float* thr  = (float*)alloc(64 * 4);

    hipMemsetAsync(comp, 0, (size_t)BB * NP * 64 * 4, stream);
    hipMemsetAsync(coop, 0, (size_t)BB * NP * 64 * 4, stream);

    k_build_x<<<128, 256, 0, stream>>>(op_idx, vf, tp_tab, cs_tab, x);
    k_gemm_h<<<BB * NP, 64, 0, stream>>>(x, 15, x, 0, Wc, asrc_c, adst_c, h1, hb1, hs1, hd1);
    k_gat<<<BB * NT, 128, 0, stream>>>(hb1, hd1, hs1, adjc, dist, wd_c, comp);
    k_gemm_h<<<BB * NP, 64, 0, stream>>>(x, 15, comp, 64, Wco, asrc_co, adst_co, h2, hb2, hs2, hd2);
    k_gat<<<BB * NT, 128, 0, stream>>>(hb2, hd2, hs2, adjo, dist, wd_co, coop);
    k_wpconv<<<160, 256, 0, stream>>>(Wp, Wb);
    k_sag<<<(BB * NP * 160) / 256, 256, 0, stream>>>(time_idx, op_idx, vf, time_tab,
                                                     tp_tab, cs_tab, comp, coop, sag);
    k_proj<<<dim3((BB * NP) / 16, 4), 128, 0, stream>>>(sag, Wb, bp, xp);
    k_scores<<<BB * NP, 256, 0, stream>>>(xp, vpd, vpf, sd, sf);
    k_topk<<<64, 1024, 0, stream>>>(sd, sf, mark, smax, thr);
    k_pool<<<64, 256, 0, stream>>>(xp, sd, sf, mark, smax, thr, out);
}
